// QATMonDEQClassifier_4037269258581
// MI455X (gfx1250) — matrix operand order repaired and checked by reading, not compiled, not run on hardware
//
#include <hip/hip_runtime.h>
#include <hip/hip_bf16.h>
#include <math.h>

// ---------------- problem constants ----------------
#define N_DIM     1024
#define D_DIM     784
#define B_DIM     8192
#define OUT_DIM   10
#define ALPHA     0.5f
#define TOL       1e-5f
#define MAX_ITERS 300
#define QMAX      127.0f

typedef __attribute__((ext_vector_type(16))) __bf16 v16bf;
typedef __attribute__((ext_vector_type(8)))  __bf16 v8bf;
typedef __attribute__((ext_vector_type(8)))  float  v8f;
typedef __attribute__((ext_vector_type(4)))  int    v4i;
typedef unsigned int u32;

// gfx1250 async global->LDS DMA (ASYNCcnt) if the toolchain exposes it.
#if defined(__AMDGCN__) && __has_builtin(__builtin_amdgcn_global_load_async_to_lds_b128)
#define USE_ASYNC_LDS 1
#else
#define USE_ASYNC_LDS 0
#endif

// ---------------- workspace layout (bytes) ----------------
// NOTE: z and inj are stored TRANSPOSED: [B][N]. This makes both GEMM operands
// K-contiguous in memory (A = zT rows, B = W rows), eliminating all transposes.
static const size_t OFF_INJ  = 0;                                            // f32 [B][N]      32 MB
static const size_t OFF_ZA   = OFF_INJ + (size_t)N_DIM * B_DIM * 4;          // bf16 [B][N]     16 MB
static const size_t OFF_ZB   = OFF_ZA  + (size_t)N_DIM * B_DIM * 2;          // bf16 [B][N]     16 MB
static const size_t OFF_W    = OFF_ZB  + (size_t)N_DIM * B_DIM * 2;          // f32 [N][N]       4 MB
static const size_t OFF_WQ   = OFF_W   + (size_t)N_DIM * N_DIM * 4;          // bf16 [N][N]      2 MB
static const size_t OFF_UQ   = OFF_WQ  + (size_t)N_DIM * N_DIM * 2;          // bf16 [N][D]    1.6 MB
static const size_t OFF_BQ   = OFF_UQ  + (size_t)N_DIM * D_DIM * 2;          // f32 [N]
static const size_t OFF_CTRL = ((OFF_BQ + (size_t)N_DIM * 4 + 255) & ~(size_t)255); // u32[8]
static const size_t OFF_PART = OFF_CTRL + 256;                               // f32 [512*2]
// ctrl[0]=absmax(U), ctrl[1]=absmax(b), ctrl[2]=absmax(W), ctrl[3]=flag, ctrl[4]=parity

// ---------------- fragment helpers (wave32 WMMA bf16 layouts) ----------------
__device__ __forceinline__ v16bf cat16(v8bf lo, v8bf hi) {
  return __builtin_shufflevector(lo, hi, 0,1,2,3,4,5,6,7,8,9,10,11,12,13,14,15);
}
// A (16x32): lane h=lane>>4 holds K=h*8..h*8+7 then K=16+h*8..; row = lane&15.
__device__ __forceinline__ v16bf load_a_frag(const __bf16* s, int row, int hv) {
  const __bf16* p = s + row * 40 + hv * 8;
  return cat16(*(const v8bf*)p, *(const v8bf*)(p + 16));
}
// B (32x16): lane h=lane>>4 holds K=h*8..h*8+7 then K=16+h*8..; col = lane&15.
__device__ __forceinline__ v16bf load_b_frag(const __bf16* s, int col, int hv) {
  const __bf16* p = s + col * 40 + hv * 8;
  return cat16(*(const v8bf*)p, *(const v8bf*)(p + 16));
}

// Stage a [128 rows][32 K] bf16 tile (row pitch 1024) into LDS (stride 40).
// 512 16-byte chunks; 256 threads -> 2 chunks/thread.
__device__ __forceinline__ void stage_tile(const __bf16* gbase, __bf16* lbuf, int kb, int tid) {
#pragma unroll
  for (int c = tid; c < 512; c += 256) {
    const int row = c >> 2, kc = (c & 3) * 8;
    const __bf16* g = gbase + (size_t)row * 1024 + kb + kc;
    __bf16* l = lbuf + row * 40 + kc;
#if USE_ASYNC_LDS
    __builtin_amdgcn_global_load_async_to_lds_b128(
        (__attribute__((address_space(1))) v4i*)g,
        (__attribute__((address_space(3))) v4i*)l, 0, 0);
#else
    *(v8bf*)l = *(const v8bf*)g;
#endif
  }
}
__device__ __forceinline__ void wait_async_lds() {
#if USE_ASYNC_LDS
#if __has_builtin(__builtin_amdgcn_s_wait_asynccnt)
  __builtin_amdgcn_s_wait_asynccnt(0);
#else
  asm volatile("s_wait_asynccnt 0x0" ::: "memory");
#endif
#endif
}

// ---------------- init: zero z0, control words ----------------
__global__ void init_k(uint4* __restrict__ z4, u32 n4, u32* __restrict__ ctrl) {
  for (u32 i = blockIdx.x * blockDim.x + threadIdx.x; i < n4; i += gridDim.x * blockDim.x)
    z4[i] = make_uint4(0u, 0u, 0u, 0u);
  if (blockIdx.x == 0 && threadIdx.x < 8) ctrl[threadIdx.x] = 0u;
}

// ---------------- abs-max (deterministic final value) ----------------
__global__ void absmax_k(const float* __restrict__ p, int n, u32* __restrict__ out) {
  float m = 0.f;
  for (int i = blockIdx.x * blockDim.x + threadIdx.x; i < n; i += gridDim.x * blockDim.x)
    m = fmaxf(m, fabsf(p[i]));
  __shared__ float red[256];
  red[threadIdx.x] = m;
  __syncthreads();
  for (int s = 128; s > 0; s >>= 1) {
    if (threadIdx.x < s) red[threadIdx.x] = fmaxf(red[threadIdx.x], red[threadIdx.x + s]);
    __syncthreads();
  }
  if (threadIdx.x == 0) atomicMax(out, __float_as_uint(red[0]));
}

// ---------------- fake-quant ----------------
__global__ void quant_bf16_k(const float* __restrict__ src, __bf16* __restrict__ dst,
                             int n, const u32* __restrict__ amax_bits) {
  float s = fmaxf(__uint_as_float(*amax_bits) / QMAX, 1e-12f);
  for (int i = blockIdx.x * blockDim.x + threadIdx.x; i < n; i += gridDim.x * blockDim.x) {
    float q = rintf(src[i] / s);
    q = fminf(fmaxf(q, -QMAX), QMAX);
    dst[i] = (__bf16)(q * s);
  }
}
__global__ void quant_f32_k(const float* __restrict__ src, float* __restrict__ dst,
                            int n, const u32* __restrict__ amax_bits) {
  float s = fmaxf(__uint_as_float(*amax_bits) / QMAX, 1e-12f);
  for (int i = blockIdx.x * blockDim.x + threadIdx.x; i < n; i += gridDim.x * blockDim.x) {
    float q = rintf(src[i] / s);
    q = fminf(fmaxf(q, -QMAX), QMAX);
    dst[i] = q * s;
  }
}

// ---------------- W = (1-softplus(m)) I - A^T A + S - S^T (one-off fp32) ----------------
__global__ void build_w_k(const float* __restrict__ A, const float* __restrict__ S,
                          const float* __restrict__ m_raw, float* __restrict__ W,
                          u32* __restrict__ wmax_bits) {
  __shared__ float At[16][17];
  __shared__ float Bt[16][17];
  __shared__ float red[256];
  const int tx = threadIdx.x, ty = threadIdx.y;
  const int i = blockIdx.y * 16 + ty;
  const int j = blockIdx.x * 16 + tx;
  float acc = 0.f;
  for (int kb = 0; kb < N_DIM; kb += 16) {
    At[ty][tx] = A[(size_t)(kb + ty) * N_DIM + blockIdx.y * 16 + tx];
    Bt[ty][tx] = A[(size_t)(kb + ty) * N_DIM + blockIdx.x * 16 + tx];
    __syncthreads();
#pragma unroll
    for (int k = 0; k < 16; ++k) acc += At[k][ty] * Bt[k][tx];
    __syncthreads();
  }
  float xr = m_raw[0];
  float m  = (xr > 20.f) ? xr : log1pf(expf(xr));
  float w  = ((i == j) ? (1.f - m) : 0.f) - acc
             + S[(size_t)i * N_DIM + j] - S[(size_t)j * N_DIM + i];
  W[(size_t)i * N_DIM + j] = w;

  const int tid = ty * 16 + tx;
  red[tid] = fabsf(w);
  __syncthreads();
  for (int s = 128; s > 0; s >>= 1) {
    if (tid < s) red[tid] = fmaxf(red[tid], red[tid + s]);
    __syncthreads();
  }
  if (tid == 0) atomicMax(wmax_bits, __float_as_uint(red[0]));
}

// ---------------- injT = x @ Uq^T + bq   (WMMA bf16; K=784, chunk-uniform pad) ----------------
__global__ __launch_bounds__(256)
void inj_k(const __bf16* __restrict__ Uq, const float* __restrict__ x,
           const float* __restrict__ bq, float* __restrict__ injT) {
  __shared__ __bf16 sA[128 * 40];   // [batch row][K=d]  from x (converted)
  __shared__ __bf16 sB[128 * 40];   // [n col][K=d]      from Uq rows (natural)
  const int tid  = threadIdx.x;
  const int lane = tid & 31, wid = tid >> 5;
  const int wm = wid >> 1, wn = wid & 1;
  const int ln = lane & 15, hv = lane >> 4;
  const int mb = blockIdx.y * 128;   // batch tile
  const int nb = blockIdx.x * 128;   // N-out tile

  v8f acc[2][4];
#pragma unroll
  for (int tm = 0; tm < 2; ++tm)
#pragma unroll
    for (int tn = 0; tn < 4; ++tn)
#pragma unroll
      for (int e = 0; e < 8; ++e) acc[tm][tn][e] = 0.f;

  for (int kb = 0; kb < 800; kb += 32) {
    { // A: x fp32 -> bf16, [row][K]; 784 % 16 == 0 so chunks are all-valid or all-pad
      const int row = tid >> 1, seg = (tid & 1) * 16;
      __bf16* dst = &sA[row * 40 + seg];
      if (kb + seg + 16 <= D_DIM) {
        const float4* g4 = (const float4*)(x + (size_t)(mb + row) * D_DIM + kb + seg);
        float4 f0 = g4[0], f1 = g4[1], f2 = g4[2], f3 = g4[3];
        v8bf o0, o1;
        o0[0]=(__bf16)f0.x; o0[1]=(__bf16)f0.y; o0[2]=(__bf16)f0.z; o0[3]=(__bf16)f0.w;
        o0[4]=(__bf16)f1.x; o0[5]=(__bf16)f1.y; o0[6]=(__bf16)f1.z; o0[7]=(__bf16)f1.w;
        o1[0]=(__bf16)f2.x; o1[1]=(__bf16)f2.y; o1[2]=(__bf16)f2.z; o1[3]=(__bf16)f2.w;
        o1[4]=(__bf16)f3.x; o1[5]=(__bf16)f3.y; o1[6]=(__bf16)f3.z; o1[7]=(__bf16)f3.w;
        *(v8bf*)dst = o0; *(v8bf*)(dst + 8) = o1;
      } else {
        v8bf zz;
#pragma unroll
        for (int e = 0; e < 8; ++e) zz[e] = (__bf16)0.f;
        *(v8bf*)dst = zz; *(v8bf*)(dst + 8) = zz;
      }
    }
    { // B: Uq rows, [n][K]
      const int row = tid >> 1, seg = (tid & 1) * 16;
      __bf16* dst = &sB[row * 40 + seg];
      if (kb + seg + 16 <= D_DIM) {
        const __bf16* g = Uq + (size_t)(nb + row) * D_DIM + kb + seg;
        *(v8bf*)dst = *(const v8bf*)g;
        *(v8bf*)(dst + 8) = *(const v8bf*)(g + 8);
      } else {
        v8bf zz;
#pragma unroll
        for (int e = 0; e < 8; ++e) zz[e] = (__bf16)0.f;
        *(v8bf*)dst = zz; *(v8bf*)(dst + 8) = zz;
      }
    }
    __syncthreads();

    v16bf af[2];
#pragma unroll
    for (int tm = 0; tm < 2; ++tm)
      af[tm] = load_a_frag(sA, wm * 32 + tm * 16 + ln, hv);
#pragma unroll
    for (int tn = 0; tn < 4; ++tn) {
      v16bf bf = load_b_frag(sB, wn * 64 + tn * 16 + ln, hv);
#pragma unroll
      for (int tm = 0; tm < 2; ++tm)
        acc[tm][tn] = __builtin_amdgcn_wmma_f32_16x16x32_bf16(
            false, af[tm], false, bf, (short)0, acc[tm][tn], false, false);
    }
    __syncthreads();
  }

#pragma unroll
  for (int tm = 0; tm < 2; ++tm)
#pragma unroll
    for (int tn = 0; tn < 4; ++tn)
#pragma unroll
      for (int r = 0; r < 8; ++r) {
        int mg = mb + wm * 32 + tm * 16 + hv * 8 + r;   // batch
        int ng = nb + wn * 64 + tn * 16 + ln;           // N-out
        injT[(size_t)mg * N_DIM + ng] = acc[tm][tn][r] + bq[ng];
      }
}

// ---------------- iteration: znT = relu((1-a) zT + a (zT W^T + injT)) ----------------
// Double-buffered LDS; async global->LDS DMA overlaps the WMMAs when available.
__global__ __launch_bounds__(256)
void iter_k(const __bf16* __restrict__ Wq, const __bf16* __restrict__ zin,
            const float* __restrict__ injT, __bf16* __restrict__ zout,
            float* __restrict__ partials, const u32* __restrict__ ctrl) {
  if (ctrl[3]) return;   // converged earlier: parity latched in ctrl[4]

  __shared__ __bf16 sA[2][128 * 40];   // zT tile [batch][K]
  __shared__ __bf16 sB[2][128 * 40];   // W  tile [n][K]  (B = W^T columns = W rows)
  __shared__ float  redd[256];
  __shared__ float  redn[256];

  const int tid  = threadIdx.x;
  const int lane = tid & 31, wid = tid >> 5;
  const int wm = wid >> 1, wn = wid & 1;
  const int ln = lane & 15, hv = lane >> 4;
  const int mb = blockIdx.y * 128;   // batch tile
  const int nb = blockIdx.x * 128;   // N-out tile
  const __bf16* zrow = zin + (size_t)mb * N_DIM;
  const __bf16* wrow = Wq  + (size_t)nb * N_DIM;

  v8f acc[2][4];
#pragma unroll
  for (int tm = 0; tm < 2; ++tm)
#pragma unroll
    for (int tn = 0; tn < 4; ++tn)
#pragma unroll
      for (int e = 0; e < 8; ++e) acc[tm][tn][e] = 0.f;

  stage_tile(zrow, sA[0], 0, tid);
  stage_tile(wrow, sB[0], 0, tid);
  wait_async_lds();
  __syncthreads();

  int cur = 0;
  for (int kb = 0; kb < N_DIM; kb += 32) {
    const bool have_next = (kb + 32) < N_DIM;
    if (have_next) {                       // DMA next tile while computing current
      stage_tile(zrow, sA[cur ^ 1], kb + 32, tid);
      stage_tile(wrow, sB[cur ^ 1], kb + 32, tid);
    }
    v16bf af[2];
#pragma unroll
    for (int tm = 0; tm < 2; ++tm)
      af[tm] = load_a_frag(sA[cur], wm * 32 + tm * 16 + ln, hv);
#pragma unroll
    for (int tn = 0; tn < 4; ++tn) {
      v16bf bf = load_b_frag(sB[cur], wn * 64 + tn * 16 + ln, hv);
#pragma unroll
      for (int tm = 0; tm < 2; ++tm)
        acc[tm][tn] = __builtin_amdgcn_wmma_f32_16x16x32_bf16(
            false, af[tm], false, bf, (short)0, acc[tm][tn], false, false);
    }
    if (have_next) {
      wait_async_lds();
      __syncthreads();
      cur ^= 1;
    }
  }

  // fused epilogue + deterministic per-block norm partials
  float d2 = 0.f, n2 = 0.f;
#pragma unroll
  for (int tm = 0; tm < 2; ++tm)
#pragma unroll
    for (int tn = 0; tn < 4; ++tn)
#pragma unroll
      for (int r = 0; r < 8; ++r) {
        int mg = mb + wm * 32 + tm * 16 + hv * 8 + r;
        int ng = nb + wn * 64 + tn * 16 + ln;
        size_t idx = (size_t)mg * N_DIM + ng;
        float zo = (float)zin[idx];
        float zn = (1.f - ALPHA) * zo + ALPHA * (acc[tm][tn][r] + injT[idx]);
        zn = fmaxf(zn, 0.f);
        zout[idx] = (__bf16)zn;
        float df = zn - zo;
        d2 += df * df;
        n2 += zo * zo;
      }
  redd[tid] = d2; redn[tid] = n2;
  __syncthreads();
  for (int s = 128; s > 0; s >>= 1) {
    if (tid < s) { redd[tid] += redd[tid + s]; redn[tid] += redn[tid + s]; }
    __syncthreads();
  }
  if (tid == 0) {
    int bid = blockIdx.y * gridDim.x + blockIdx.x;
    partials[2 * bid]     = redd[0];
    partials[2 * bid + 1] = redn[0];
  }
}

// ---------------- deterministic convergence check ----------------
__global__ void check_k(const float* __restrict__ part, int nblocks,
                        u32* __restrict__ ctrl, int next_parity) {
  __shared__ float sd[256], sn[256];
  float d = 0.f, n = 0.f;
  for (int i = threadIdx.x; i < nblocks; i += 256) { d += part[2 * i]; n += part[2 * i + 1]; }
  sd[threadIdx.x] = d; sn[threadIdx.x] = n;
  __syncthreads();
  for (int s = 128; s > 0; s >>= 1) {
    if (threadIdx.x < s) { sd[threadIdx.x] += sd[threadIdx.x + s]; sn[threadIdx.x] += sn[threadIdx.x + s]; }
    __syncthreads();
  }
  if (threadIdx.x == 0) {
    float err = sqrtf(sd[0]) / (sqrtf(sn[0]) + 1e-12f);
    if (err < TOL && ctrl[3] == 0u) { ctrl[3] = 1u; ctrl[4] = (u32)next_parity; }
  }
}

// ---------------- logits = zT @ out_W^T + out_b  (zT rows are contiguous) ----------------
__global__ __launch_bounds__(256)
void logits_k(const __bf16* __restrict__ zA, const __bf16* __restrict__ zB,
              const float* __restrict__ outW, const float* __restrict__ outb,
              const u32* __restrict__ ctrl, float* __restrict__ out) {
  __shared__ float swgt[OUT_DIM * N_DIM];
  for (int i = threadIdx.x; i < OUT_DIM * N_DIM; i += 256) swgt[i] = outW[i];
  __syncthreads();
  const __bf16* z = ctrl[4] ? zB : zA;
  const int lane = threadIdx.x & 31, wid = threadIdx.x >> 5;
  const int j = blockIdx.x * 8 + wid;   // batch row
  float acc[OUT_DIM];
#pragma unroll
  for (int o = 0; o < OUT_DIM; ++o) acc[o] = 0.f;
  for (int i = lane; i < N_DIM; i += 32) {
    float zv = (float)z[(size_t)j * N_DIM + i];
#pragma unroll
    for (int o = 0; o < OUT_DIM; ++o) acc[o] += zv * swgt[o * N_DIM + i];
  }
#pragma unroll
  for (int o = 0; o < OUT_DIM; ++o)
#pragma unroll
    for (int off = 16; off > 0; off >>= 1) acc[o] += __shfl_xor(acc[o], off, 32);
  if (lane == 0)
#pragma unroll
    for (int o = 0; o < OUT_DIM; ++o) out[(size_t)j * OUT_DIM + o] = acc[o] + outb[o];
}

// ---------------- launcher ----------------
extern "C" void kernel_launch(void* const* d_in, const int* in_sizes, int n_in,
                              void* d_out, int out_size, void* d_ws, size_t ws_size,
                              hipStream_t stream) {
  const float* A     = (const float*)d_in[0];
  const float* S     = (const float*)d_in[1];
  const float* m_raw = (const float*)d_in[2];
  const float* U     = (const float*)d_in[3];
  const float* b     = (const float*)d_in[4];
  const float* x     = (const float*)d_in[5];
  const float* outW  = (const float*)d_in[6];
  const float* outb  = (const float*)d_in[7];

  char* ws = (char*)d_ws;
  float*  injT = (float*)(ws + OFF_INJ);
  __bf16* zA   = (__bf16*)(ws + OFF_ZA);
  __bf16* zB   = (__bf16*)(ws + OFF_ZB);
  float*  Wf   = (float*)(ws + OFF_W);
  __bf16* Wq   = (__bf16*)(ws + OFF_WQ);
  __bf16* Uq   = (__bf16*)(ws + OFF_UQ);
  float*  bq   = (float*)(ws + OFF_BQ);
  u32*    ctrl = (u32*)(ws + OFF_CTRL);
  float*  part = (float*)(ws + OFF_PART);

  init_k<<<2048, 256, 0, stream>>>((uint4*)zA, (u32)((size_t)N_DIM * B_DIM * 2 / 16), ctrl);

  absmax_k<<<512, 256, 0, stream>>>(U, N_DIM * D_DIM, ctrl + 0);
  absmax_k<<<8,   256, 0, stream>>>(b, N_DIM,          ctrl + 1);
  quant_bf16_k<<<784, 256, 0, stream>>>(U, Uq, N_DIM * D_DIM, ctrl + 0);
  quant_f32_k<<<4,    256, 0, stream>>>(b, bq, N_DIM,          ctrl + 1);

  build_w_k<<<dim3(64, 64), dim3(16, 16), 0, stream>>>(A, S, m_raw, Wf, ctrl + 2);
  quant_bf16_k<<<1024, 256, 0, stream>>>(Wf, Wq, N_DIM * N_DIM, ctrl + 2);

  // injT = x @ Uq^T + bq
  inj_k<<<dim3(8, 64), 256, 0, stream>>>(Uq, x, bq, injT);

  // fixed-point iterations with device-side deterministic early-exit
  __bf16* zb[2] = { zA, zB };
  for (int t = 0; t < MAX_ITERS; ++t) {
    iter_k<<<dim3(8, 64), 256, 0, stream>>>(Wq, zb[t & 1], injT, zb[(t + 1) & 1], part, ctrl);
    check_k<<<1, 256, 0, stream>>>(part, 512, ctrl, (t + 1) & 1);
  }

  logits_k<<<1024, 256, 0, stream>>>(zA, zB, outW, outb, ctrl, (float*)d_out);

  (void)in_sizes; (void)n_in; (void)out_size; (void)ws_size;
}
